// GlobalModel_NodeOnly_A_26302379720749
// MI455X (gfx1250) — matrix operand order repaired and checked by reading, not compiled, not run on hardware
//
#include <hip/hip_runtime.h>
#include <hip/hip_bf16.h>

// ---------------------------------------------------------------------------
// Problem constants (match reference): N=1e6 nodes, B=4096 graphs,
// F_X=F_U=128, H=64. Dominant cost: x@Wk ([1M x 128]x[128 x 64]) fused with
// the sigmoid gate and the segment-sum, reading x from HBM exactly once.
// ---------------------------------------------------------------------------
#define NN 1000000
#define BB 4096
#define FX 128
#define FU 128
#define HH 64

typedef __attribute__((ext_vector_type(16))) __bf16 v16bf;
typedef __attribute__((ext_vector_type(8)))  float  v8f;

// ------------------------- zero workspace ----------------------------------
__global__ void zero_kernel(float* __restrict__ p, int n) {
  int i = blockIdx.x * blockDim.x + threadIdx.x;
  if (i < n) p[i] = 0.0f;
}

// ------------------------- q = u @ Wq + bq  (tiny, f32) --------------------
__global__ void q_kernel(const float* __restrict__ u, const float* __restrict__ Wq,
                         const float* __restrict__ bq, float* __restrict__ q) {
  int idx = blockIdx.x * blockDim.x + threadIdx.x;
  if (idx >= BB * HH) return;
  int b = idx >> 6;          // graph
  int h = idx & 63;          // head dim
  float s = bq[h];
  const float* ur = u + b * FU;
#pragma unroll 8
  for (int f = 0; f < FU; ++f) s = fmaf(ur[f], Wq[f * HH + h], s);
  q[idx] = s;
}

// ---------------------------------------------------------------------------
// Fused: k = x@Wk + bk (bf16 WMMA, f32 accum), gate = sigmoid(k . q[batch]),
// x_agg[g] += gate * x   (run-length compressed atomics over sorted batch).
// One wave32 per 16-node tile; 8 waves (128 nodes) per workgroup.
// ---------------------------------------------------------------------------
__global__ void __launch_bounds__(256) gate_scatter_kernel(
    const float* __restrict__ x, const long long* __restrict__ batch,
    const float* __restrict__ Wk, const float* __restrict__ bk,
    const float* __restrict__ q, float* __restrict__ xagg) {
  // Wk transposed+converted to bf16 in LDS: lwk[n][k], n in [0,64), k in [0,128)
  __shared__ __align__(32) __bf16 lwk[HH * FX];

  int tid = threadIdx.x;
  for (int i = tid; i < FX * HH; i += 256) {
    int k = i >> 6;          // Wk is [128][64] row-major
    int n = i & 63;
    lwk[n * FX + k] = (__bf16)Wk[i];
  }
  __syncthreads();

  int lane = tid & 31;
  int wave = tid >> 5;
  int lo = lane & 15;        // row within tile / column within n-tile
  int hi = lane >> 4;        // half-wave selector (K-block split)
  int base = blockIdx.x * 128 + wave * 16;   // first node of this wave's tile

  // Clamp so padded tail lanes still load valid memory (EXEC stays all-ones
  // for WMMA); their gate results are never stored.
  int nodeA = base + lo;
  if (nodeA >= NN) nodeA = NN - 1;
  const float4* xr = (const float4*)(x + (size_t)nodeA * FX);

  v8f z = {0.f, 0.f, 0.f, 0.f, 0.f, 0.f, 0.f, 0.f};
  v8f acc[4] = {z, z, z, z};   // k tile: acc[t][r] -> M=r+8*hi, n=t*16+lo

  // ---- k = x @ Wk via v_wmma_f32_16x16x32_bf16: 4 K-chunks x 4 N-tiles ----
#pragma unroll
  for (int c = 0; c < 4; ++c) {
    // A operand (16-bit, 16x32), contiguous k order used here, not AMD's table order: lanes 0-15 hold K = c*32 + {0..15},
    // lanes 16-31 hold K = c*32 + {16..31}; element i maps to VGPR i/2.
    int f4 = c * 8 + hi * 4;                      // float4 index of K-block
    float4 p0 = xr[f4 + 0], p1 = xr[f4 + 1];      // K = koff .. koff+7
    float4 p2 = xr[f4 + 2], p3 = xr[f4 + 3];      // K = koff+8 .. koff+15
    v16bf a;
    a[0]  = (__bf16)p0.x; a[1]  = (__bf16)p0.y; a[2]  = (__bf16)p0.z; a[3]  = (__bf16)p0.w;
    a[4]  = (__bf16)p1.x; a[5]  = (__bf16)p1.y; a[6]  = (__bf16)p1.z; a[7]  = (__bf16)p1.w;
    a[8]  = (__bf16)p2.x; a[9]  = (__bf16)p2.y; a[10] = (__bf16)p2.z; a[11] = (__bf16)p2.w;
    a[12] = (__bf16)p3.x; a[13] = (__bf16)p3.y; a[14] = (__bf16)p3.z; a[15] = (__bf16)p3.w;
#pragma unroll
    for (int t = 0; t < 4; ++t) {
      // B operand (32x16), same contiguous k order as A: lanes 0-15 -> K=c*32..+15, lanes 16-31 -> +16..31,
      // column n = t*16+lo. Contiguous 32B from transposed LDS image of Wk.
      const v16bf bm = *(const v16bf*)&lwk[(t * 16 + lo) * FX + c * 32 + hi * 16];
      acc[t] = __builtin_amdgcn_wmma_f32_16x16x32_bf16(
          false, a, false, bm, (short)0, acc[t], false, false);
    }
  }

  // ---- gate[m] = sigmoid( (k[m]+bk) . q[batch[m]] ) -----------------------
  int g_lane = (int)batch[nodeA];        // lane m (and m+16) holds graph of row m
  float bkv[4];
#pragma unroll
  for (int t = 0; t < 4; ++t) bkv[t] = bk[t * 16 + lo];

  float gate[8];                         // gate for M = r + 8*hi
#pragma unroll
  for (int r = 0; r < 8; ++r) {
    int g = __shfl(g_lane, r + hi * 8, 32);
    float s = 0.f;
#pragma unroll
    for (int t = 0; t < 4; ++t) {
      float kv = acc[t][r] + bkv[t];
      s = fmaf(kv, q[g * HH + t * 16 + lo], s);
    }
    // reduce over the 16 lanes of this half (n dimension)
    s += __shfl_xor(s, 1, 32);
    s += __shfl_xor(s, 2, 32);
    s += __shfl_xor(s, 4, 32);
    s += __shfl_xor(s, 8, 32);
    gate[r] = 1.0f / (1.0f + __expf(-s));
  }

  // ---- x_agg[batch[m]] += gate[m] * x[m]  (sorted batch -> RLE atomics) ---
  // Lane covers features [lane*4, lane*4+4). x rows are L0/L2-hot (just read).
  int gprev = -1;
  float r0 = 0.f, r1 = 0.f, r2 = 0.f, r3 = 0.f;
#pragma unroll
  for (int m = 0; m < 16; ++m) {
    int node = base + m;
    if (node < NN) {
      int g = __shfl(g_lane, m, 32);
      float am = __shfl(gate[m & 7], (m & 8) ? 16 : 0, 32);
      float4 xv = ((const float4*)(x + (size_t)node * FX))[lane];
      if (g != gprev) {
        if (gprev >= 0) {
          float* dst = xagg + (size_t)gprev * FX + lane * 4;
          atomicAdd(dst + 0, r0); atomicAdd(dst + 1, r1);
          atomicAdd(dst + 2, r2); atomicAdd(dst + 3, r3);
        }
        gprev = g; r0 = r1 = r2 = r3 = 0.f;
      }
      r0 = fmaf(am, xv.x, r0); r1 = fmaf(am, xv.y, r1);
      r2 = fmaf(am, xv.z, r2); r3 = fmaf(am, xv.w, r3);
    }
  }
  if (gprev >= 0) {
    float* dst = xagg + (size_t)gprev * FX + lane * 4;
    atomicAdd(dst + 0, r0); atomicAdd(dst + 1, r1);
    atomicAdd(dst + 2, r2); atomicAdd(dst + 3, r3);
  }
}

// --------------- out = concat([x_agg, u]) @ Wu + bu  (tiny, f32) -----------
__global__ void out_kernel(const float* __restrict__ xagg, const float* __restrict__ u,
                           const float* __restrict__ Wu, const float* __restrict__ bu,
                           float* __restrict__ out) {
  int idx = blockIdx.x * blockDim.x + threadIdx.x;
  if (idx >= BB * FU) return;
  int b = idx >> 7;
  int j = idx & 127;
  float s = bu[j];
  const float* xa = xagg + b * FX;
  const float* ur = u + b * FU;
#pragma unroll 4
  for (int f = 0; f < FX; ++f) s = fmaf(xa[f], Wu[f * FU + j], s);
#pragma unroll 4
  for (int f = 0; f < FU; ++f) s = fmaf(ur[f], Wu[(FX + f) * FU + j], s);
  out[idx] = s;
}

// ---------------------------------------------------------------------------
extern "C" void kernel_launch(void* const* d_in, const int* in_sizes, int n_in,
                              void* d_out, int out_size, void* d_ws, size_t ws_size,
                              hipStream_t stream) {
  (void)in_sizes; (void)n_in; (void)out_size; (void)ws_size;
  const float*     x     = (const float*)d_in[0];
  // d_in[1] = edge_index (unused), d_in[2] = e (unused)
  const float*     u     = (const float*)d_in[3];
  const long long* batch = (const long long*)d_in[4];
  const float*     Wk    = (const float*)d_in[5];
  const float*     bk    = (const float*)d_in[6];
  const float*     Wq    = (const float*)d_in[7];
  const float*     bq    = (const float*)d_in[8];
  const float*     Wu    = (const float*)d_in[9];
  const float*     bu    = (const float*)d_in[10];
  float* out  = (float*)d_out;

  float* xagg = (float*)d_ws;          // [B, F_X]  = 2 MB
  float* q    = xagg + BB * FX;        // [B, H]    = 1 MB

  zero_kernel<<<(BB * FX + 255) / 256, 256, 0, stream>>>(xagg, BB * FX);
  q_kernel<<<(BB * HH + 255) / 256, 256, 0, stream>>>(u, Wq, bq, q);
  gate_scatter_kernel<<<(NN + 127) / 128, 256, 0, stream>>>(x, batch, Wk, bk, q, xagg);
  out_kernel<<<(BB * FU + 255) / 256, 256, 0, stream>>>(xagg, u, Wu, bu, out);
}
